// DeltaNet_25881472926370
// MI455X (gfx1250) — matrix operand order repaired and checked by reading, not compiled, not run on hardware
//
#include <hip/hip_runtime.h>

#define LAYERS 2
#define DIM    1024
#define NH     8
#define HD     128
#define SEQ    2048
#define BATCH  4
#define INTERD 2816
#define ROWS   (BATCH*SEQ)      // 8192
#define EPSF   1e-6f

typedef __bf16 bf16_t;
typedef __attribute__((ext_vector_type(16))) __bf16 v16bf;
typedef __attribute__((ext_vector_type(8)))  __bf16 v8bf;
typedef __attribute__((ext_vector_type(8)))  float  v8f;
typedef __attribute__((address_space(3))) float lds_f32;

static __device__ __forceinline__ bf16_t f2bf(float f) {
    unsigned u = __builtin_bit_cast(unsigned, f);
    unsigned r = u + 0x7FFFu + ((u >> 16) & 1u);   // round-to-nearest-even
    unsigned short h = (unsigned short)(r >> 16);
    return __builtin_bit_cast(bf16_t, h);
}
static __device__ __forceinline__ float bf2f(bf16_t b) {
    unsigned short h = __builtin_bit_cast(unsigned short, b);
    unsigned u = ((unsigned)h) << 16;
    return __builtin_bit_cast(float, u);
}
// fast sigmoid: v_exp + v_rcp instead of full-precision divide sequence
static __device__ __forceinline__ float sigm(float x) {
    return __builtin_amdgcn_rcpf(1.0f + __expf(-x));
}
static __device__ __forceinline__ float siluf(float x) { return x * sigm(x); }

// CDNA5 async global->LDS copy (ASYNCcnt-tracked), per-lane 4 bytes.
static __device__ __forceinline__ void async_load_f32(lds_f32* dst, const float* src) {
    asm volatile("global_load_async_to_lds_b32 %0, %1, off"
                 :: "v"(dst), "v"(src) : "memory");
}
static __device__ __forceinline__ void wait_async0() {
    asm volatile("s_wait_asynccnt 0" ::: "memory");
}

// ---------------------------------------------------------------------------
// A-operand fragment (contiguous k order, B loads match): per lane, K = {kb..kb+15}
// ---------------------------------------------------------------------------
static __device__ __forceinline__ v16bf load_a_frag(const bf16_t* p) {
    v8bf lo = *(const v8bf*)(p);
    v8bf hi = *(const v8bf*)(p + 8);
    return __builtin_shufflevector(lo, hi, 0,1,2,3,4,5,6,7,8,9,10,11,12,13,14,15);
}

// ---------------------------------------------------------------------------
// GEMM: C[M,N](f32) (op)= A[M,K](bf16, row-major) * Bt[N,K](bf16, row-major)^T
// One wave per 64x64 C tile (4x4 WMMA accumulators, ~32 FLOP/byte).
// EPI: 0 = silu->store, 1 = add into C (residual).
// ---------------------------------------------------------------------------
template<int EPI>
__global__ __launch_bounds__(32) void gemm_wmma(const bf16_t* __restrict__ A,
                                                const bf16_t* __restrict__ Bt,
                                                float* __restrict__ C,
                                                int N, int K)
{
    const int l  = threadIdx.x;
    const int lr = l & 15, lh = l >> 4;
    const int m0 = blockIdx.y * 64, n0 = blockIdx.x * 64;

    v8f acc[4][4];
#pragma unroll
    for (int i = 0; i < 4; ++i)
#pragma unroll
        for (int j = 0; j < 4; ++j) acc[i][j] = (v8f){};

    const bf16_t* Ab = A  + (size_t)(m0 + lr) * K;
    const bf16_t* Bb = Bt + (size_t)(n0 + lr) * K;

    for (int kk = 0; kk < K; kk += 32) {
        v16bf af[4], bfr[4];
#pragma unroll
        for (int i = 0; i < 4; ++i)
            af[i] = load_a_frag(Ab + (size_t)(i * 16) * K + kk + lh * 16);
#pragma unroll
        for (int j = 0; j < 4; ++j)
            bfr[j] = *(const v16bf*)(Bb + (size_t)(j * 16) * K + kk + lh * 16);
#pragma unroll
        for (int i = 0; i < 4; ++i)
#pragma unroll
            for (int j = 0; j < 4; ++j)
                acc[i][j] = __builtin_amdgcn_wmma_f32_16x16x32_bf16(
                    false, af[i], false, bfr[j], (short)0, acc[i][j], false, false);
    }

    const int col0 = n0 + lr;
    const int rb   = m0 + lh * 8;
#pragma unroll
    for (int i = 0; i < 4; ++i) {
#pragma unroll
        for (int j = 0; j < 8; ++j) {
            size_t r = (size_t)(rb + i * 16 + j) * N;
#pragma unroll
            for (int jj = 0; jj < 4; ++jj) {
                float v = acc[i][jj][j];
                if (EPI == 0) C[r + col0 + jj * 16]  = siluf(v);
                else          C[r + col0 + jj * 16] += v;
            }
        }
    }
}

// ---------------------------------------------------------------------------
// Fused SwiGLU GEMM: Z[M,INTER](bf16) = silu(A*Wg_g^T) * (A*Wg_y^T)
// One wave per 64x32 Z tile, dual accumulators share the 4 A fragments.
// ---------------------------------------------------------------------------
__global__ __launch_bounds__(32) void gemm_glu(const bf16_t* __restrict__ A,
                                               const bf16_t* __restrict__ Wgt,
                                               bf16_t* __restrict__ Z)
{
    const int K = DIM;
    const int l  = threadIdx.x;
    const int lr = l & 15, lh = l >> 4;
    const int m0 = blockIdx.y * 64, n0 = blockIdx.x * 32;

    v8f accg[4][2], accy[4][2];
#pragma unroll
    for (int i = 0; i < 4; ++i)
#pragma unroll
        for (int j = 0; j < 2; ++j) { accg[i][j] = (v8f){}; accy[i][j] = (v8f){}; }

    const bf16_t* Ab = A + (size_t)(m0 + lr) * K;
    const bf16_t* Gb = Wgt + (size_t)(n0 + lr) * K;
    const bf16_t* Yb = Wgt + (size_t)(INTERD + n0 + lr) * K;

    for (int kk = 0; kk < K; kk += 32) {
        v16bf af[4], bg[2], by[2];
#pragma unroll
        for (int i = 0; i < 4; ++i)
            af[i] = load_a_frag(Ab + (size_t)(i * 16) * K + kk + lh * 16);
#pragma unroll
        for (int j = 0; j < 2; ++j) {
            bg[j] = *(const v16bf*)(Gb + (size_t)(j * 16) * K + kk + lh * 16);
            by[j] = *(const v16bf*)(Yb + (size_t)(j * 16) * K + kk + lh * 16);
        }
#pragma unroll
        for (int i = 0; i < 4; ++i)
#pragma unroll
            for (int j = 0; j < 2; ++j) {
                accg[i][j] = __builtin_amdgcn_wmma_f32_16x16x32_bf16(
                    false, af[i], false, bg[j], (short)0, accg[i][j], false, false);
                accy[i][j] = __builtin_amdgcn_wmma_f32_16x16x32_bf16(
                    false, af[i], false, by[j], (short)0, accy[i][j], false, false);
            }
    }

    const int col0 = n0 + lr;
    const int rb   = m0 + lh * 8;
#pragma unroll
    for (int i = 0; i < 4; ++i) {
#pragma unroll
        for (int j = 0; j < 8; ++j) {
            size_t r = (size_t)(rb + i * 16 + j) * INTERD;
#pragma unroll
            for (int jj = 0; jj < 2; ++jj)
                Z[r + col0 + jj * 16] = f2bf(siluf(accg[i][jj][j]) * accy[i][jj][j]);
        }
    }
}

// ---------------------------------------------------------------------------
// Input transpose (T,B,D) f32 -> (B,T,D) f32
// ---------------------------------------------------------------------------
__global__ void transpose_in(const float* __restrict__ in, float* __restrict__ x)
{
    size_t tid = (size_t)blockIdx.x * blockDim.x + threadIdx.x;
    if (tid >= (size_t)SEQ * BATCH * DIM) return;
    int d = (int)(tid % DIM);
    int b = (int)((tid / DIM) % BATCH);
    int t = (int)(tid / ((size_t)DIM * BATCH));
    x[((size_t)b * SEQ + t) * DIM + d] = in[tid];
}

// ---------------------------------------------------------------------------
// Weight convert + transpose: W[K,N] f32 -> Wt[N,K] bf16
// ---------------------------------------------------------------------------
__global__ void convT_bf(const float* __restrict__ W, bf16_t* __restrict__ Wt, int K, int N)
{
    size_t tid = (size_t)blockIdx.x * blockDim.x + threadIdx.x;
    if (tid >= (size_t)K * N) return;
    int k = (int)(tid / N);
    int n = (int)(tid % N);
    Wt[(size_t)n * K + k] = f2bf(W[tid]);
}

// ---------------------------------------------------------------------------
// Row RMSNorm (rows of DIM), bf16 output
// ---------------------------------------------------------------------------
__global__ __launch_bounds__(256) void rmsnorm_bf(const float* __restrict__ x,
                                                  const float* __restrict__ w,
                                                  bf16_t* __restrict__ out)
{
    __shared__ float red[256];
    const int r = blockIdx.x, tid = threadIdx.x;
    float vals[4];
    float acc = 0.f;
#pragma unroll
    for (int i = 0; i < 4; ++i) {
        vals[i] = x[(size_t)r * DIM + tid + i * 256];
        acc += vals[i] * vals[i];
    }
    red[tid] = acc;
    __syncthreads();
    for (int st = 128; st > 0; st >>= 1) { if (tid < st) red[tid] += red[tid + st]; __syncthreads(); }
    float scale = rsqrtf(red[0] / (float)DIM + EPSF);
#pragma unroll
    for (int i = 0; i < 4; ++i) {
        int d = tid + i * 256;
        out[(size_t)r * DIM + d] = f2bf(vals[i] * scale * w[d]);
    }
}

// ---------------------------------------------------------------------------
// Final RMSNorm + transpose back to (T,B,D) f32 output
// ---------------------------------------------------------------------------
__global__ __launch_bounds__(256) void final_norm_out(const float* __restrict__ x,
                                                      const float* __restrict__ w,
                                                      float* __restrict__ out)
{
    __shared__ float red[256];
    const int r = blockIdx.x, tid = threadIdx.x;
    const int b = r / SEQ, t = r % SEQ;
    float vals[4];
    float acc = 0.f;
#pragma unroll
    for (int i = 0; i < 4; ++i) {
        vals[i] = x[(size_t)r * DIM + tid + i * 256];
        acc += vals[i] * vals[i];
    }
    red[tid] = acc;
    __syncthreads();
    for (int st = 128; st > 0; st >>= 1) { if (tid < st) red[tid] += red[tid + st]; __syncthreads(); }
    float scale = rsqrtf(red[0] / (float)DIM + EPSF);
#pragma unroll
    for (int i = 0; i < 4; ++i) {
        int d = tid + i * 256;
        out[((size_t)t * BATCH + b) * DIM + d] = vals[i] * scale * w[d];
    }
}

// ---------------------------------------------------------------------------
// beta = 2*sigmoid(h @ Wb), h in bf16, Wb [D,H] f32
// ---------------------------------------------------------------------------
__global__ void beta_kernel(const bf16_t* __restrict__ h, const float* __restrict__ Wb,
                            float* __restrict__ beta)
{
    int tid = blockIdx.x * blockDim.x + threadIdx.x;
    if (tid >= ROWS * NH) return;
    int r = tid / NH, hh = tid % NH;
    float acc = 0.f;
    for (int d = 0; d < DIM; ++d)
        acc += bf2f(h[(size_t)r * DIM + d]) * Wb[(size_t)d * NH + hh];
    beta[tid] = 2.f * sigm(acc);
}

// ---------------------------------------------------------------------------
// In-place L2 norm over contiguous rows of HD=128 (q/k heads)
// ---------------------------------------------------------------------------
__global__ __launch_bounds__(128) void l2norm_rows(float* __restrict__ x)
{
    __shared__ float red[128];
    const size_t r = blockIdx.x;
    const int lane = threadIdx.x;
    float v = x[r * HD + lane];
    red[lane] = v * v;
    __syncthreads();
    for (int st = 64; st > 0; st >>= 1) { if (lane < st) red[lane] += red[lane + st]; __syncthreads(); }
    x[r * HD + lane] = v * rsqrtf(red[0] + EPSF);
}

// ---------------------------------------------------------------------------
// Delta-rule recurrence. One block per (b,h). lane owns column v of S[128x128]
// kept fully in registers. q/k/v for step t+1 are prefetched into the other
// LDS buffer with GLOBAL_LOAD_ASYNC_TO_LDS_B32 (ASYNCcnt) while step t's
// ~300-FMA chain executes. Fused per-head RMSNorm, bf16 output.
// ---------------------------------------------------------------------------
__global__ __launch_bounds__(128) void delta_scan(const float* __restrict__ q,
                                                  const float* __restrict__ k,
                                                  const float* __restrict__ v,
                                                  const float* __restrict__ beta,
                                                  const float* __restrict__ onorm_w,
                                                  bf16_t* __restrict__ o_bf)
{
    const int bh = blockIdx.x;
    const int b = bh / NH, h = bh % NH;
    const int lane = threadIdx.x;   // DV column index
    __shared__ float qs[2][HD], ks[2][HD], vs[2][HD], red[HD];

    float s[HD];
#pragma unroll
    for (int i = 0; i < HD; ++i) s[i] = 0.f;
    const float w = onorm_w[lane];

    // prologue: async-fill buffer 0 with t=0
    {
        const size_t base0 = ((size_t)(b * SEQ)) * DIM + (size_t)h * HD + lane;
        async_load_f32((lds_f32*)&qs[0][lane], q + base0);
        async_load_f32((lds_f32*)&ks[0][lane], k + base0);
        async_load_f32((lds_f32*)&vs[0][lane], v + base0);
        wait_async0();
        __syncthreads();
    }

    for (int t = 0; t < SEQ; ++t) {
        const int cur = t & 1, nxt = cur ^ 1;
        const size_t base = ((size_t)(b * SEQ + t)) * DIM + (size_t)h * HD;

        if (t + 1 < SEQ) {
            const size_t basen = base + DIM + lane;   // next timestep, same (b,h)
            async_load_f32((lds_f32*)&qs[nxt][lane], q + basen);
            async_load_f32((lds_f32*)&ks[nxt][lane], k + basen);
            async_load_f32((lds_f32*)&vs[nxt][lane], v + basen);
        }

        const float bet = beta[(size_t)(b * SEQ + t) * NH + h];
        float pred = 0.f;
#pragma unroll
        for (int i = 0; i < HD; ++i) pred += s[i] * ks[cur][i];
        const float delta = bet * (vs[cur][lane] - pred);
        float o = 0.f;
#pragma unroll
        for (int i = 0; i < HD; ++i) { s[i] += ks[cur][i] * delta; o += s[i] * qs[cur][i]; }

        red[lane] = o * o;
        __syncthreads();
        for (int st = 64; st > 0; st >>= 1) { if (lane < st) red[lane] += red[lane + st]; __syncthreads(); }
        const float scale = rsqrtf(red[0] / (float)HD + EPSF);
        o_bf[base + lane] = f2bf(o * scale * w);

        wait_async0();       // prefetch of t+1 has landed (per wave)
        __syncthreads();     // visible to all waves; also guards red[] reuse
    }
}

// ---------------------------------------------------------------------------
extern "C" void kernel_launch(void* const* d_in, const int* in_sizes, int n_in,
                              void* d_out, int out_size, void* d_ws, size_t ws_size,
                              hipStream_t stream)
{
    (void)in_sizes; (void)n_in; (void)out_size; (void)ws_size;
    const float* emb = (const float*)d_in[0];
    const float* Wq  = (const float*)d_in[1];
    const float* Wk  = (const float*)d_in[2];
    const float* Wv  = (const float*)d_in[3];
    const float* Wb  = (const float*)d_in[4];
    const float* Wo  = (const float*)d_in[5];
    const float* onw = (const float*)d_in[6];
    const float* n1w = (const float*)d_in[7];
    const float* n2w = (const float*)d_in[8];
    const float* Wg  = (const float*)d_in[9];
    const float* Wd  = (const float*)d_in[10];
    const float* fnw = (const float*)d_in[11];
    float* out = (float*)d_out;

    char* ws = (char*)d_ws;
    size_t off = 0;
    auto carve = [&](size_t bytes) -> char* {
        char* p = ws + off;
        off += (bytes + 255) & ~(size_t)255;
        return p;
    };
    float*  x    = (float*) carve((size_t)ROWS * DIM * 4);
    bf16_t* hbf  = (bf16_t*)carve((size_t)ROWS * DIM * 2);
    char*   big  = carve((size_t)3 * ROWS * DIM * 4);   // qf|kf|vf, reused as z(bf16)
    float*  qf   = (float*)big;
    float*  kf   = qf + (size_t)ROWS * DIM;
    float*  vf   = kf + (size_t)ROWS * DIM;
    bf16_t* zbf  = (bf16_t*)big;
    bf16_t* obf  = (bf16_t*)carve((size_t)ROWS * DIM * 2);
    float*  beta = (float*) carve((size_t)ROWS * NH * 4);
    bf16_t* wt   = (bf16_t*)carve((size_t)(2 * INTERD) * DIM * 2);

    transpose_in<<<(SEQ * BATCH * DIM + 255) / 256, 256, 0, stream>>>(emb, x);

    for (int l = 0; l < LAYERS; ++l) {
        const float* Wq_l = Wq + (size_t)l * DIM * DIM;
        const float* Wk_l = Wk + (size_t)l * DIM * DIM;
        const float* Wv_l = Wv + (size_t)l * DIM * DIM;
        const float* Wb_l = Wb + (size_t)l * DIM * NH;
        const float* Wo_l = Wo + (size_t)l * DIM * DIM;
        const float* Wg_l = Wg + (size_t)l * DIM * 2 * INTERD;
        const float* Wd_l = Wd + (size_t)l * INTERD * DIM;

        // --- attention block ---
        rmsnorm_bf<<<ROWS, 256, 0, stream>>>(x, n1w + l * DIM, hbf);

        convT_bf<<<(DIM * DIM + 255) / 256, 256, 0, stream>>>(Wq_l, wt, DIM, DIM);
        gemm_wmma<0><<<dim3(DIM / 64, ROWS / 64), 32, 0, stream>>>(hbf, wt, qf, DIM, DIM);
        convT_bf<<<(DIM * DIM + 255) / 256, 256, 0, stream>>>(Wk_l, wt, DIM, DIM);
        gemm_wmma<0><<<dim3(DIM / 64, ROWS / 64), 32, 0, stream>>>(hbf, wt, kf, DIM, DIM);
        convT_bf<<<(DIM * DIM + 255) / 256, 256, 0, stream>>>(Wv_l, wt, DIM, DIM);
        gemm_wmma<0><<<dim3(DIM / 64, ROWS / 64), 32, 0, stream>>>(hbf, wt, vf, DIM, DIM);

        beta_kernel<<<(ROWS * NH + 255) / 256, 256, 0, stream>>>(hbf, Wb_l, beta);
        l2norm_rows<<<ROWS * NH, 128, 0, stream>>>(qf);
        l2norm_rows<<<ROWS * NH, 128, 0, stream>>>(kf);

        delta_scan<<<BATCH * NH, 128, 0, stream>>>(qf, kf, vf, beta, onw + l * HD, obf);

        convT_bf<<<(DIM * DIM + 255) / 256, 256, 0, stream>>>(Wo_l, wt, DIM, DIM);
        gemm_wmma<1><<<dim3(DIM / 64, ROWS / 64), 32, 0, stream>>>(obf, wt, x, DIM, DIM);

        // --- MLP block ---
        rmsnorm_bf<<<ROWS, 256, 0, stream>>>(x, n2w + l * DIM, hbf);
        convT_bf<<<(DIM * 2 * INTERD + 255) / 256, 256, 0, stream>>>(Wg_l, wt, DIM, 2 * INTERD);
        gemm_glu<<<dim3(INTERD / 32, ROWS / 64), 32, 0, stream>>>(hbf, wt, zbf);
        convT_bf<<<(INTERD * DIM + 255) / 256, 256, 0, stream>>>(Wd_l, wt, INTERD, DIM);
        gemm_wmma<1><<<dim3(DIM / 64, ROWS / 64), 32, 0, stream>>>(zbf, wt, x, DIM, INTERD);
    }

    final_norm_out<<<ROWS, 256, 0, stream>>>(x, fnw, out);
}
